// CLGNN_22462678958656
// MI455X (gfx1250) — compile-verified, matrix operand order checked by reading, not run on hardware
//
#include <hip/hip_runtime.h>
#include <hip/hip_fp16.h>

// ---------------------------------------------------------------------------
// CLGNN forward for MI455X (gfx1250, wave32).
// Fully fused CEMLP block: lin -> silu -> right/left -> weighted geometric
// product -> MV layernorm in ONE kernel. Activations f16 [blade][batch][chanP];
// all GEMMs via v_wmma_f32_16x16x32_f16 (f32 accum); A-tiles staged into LDS
// with gfx1250 async global->LDS copies; intermediates live in LDS.
// ---------------------------------------------------------------------------

typedef __attribute__((ext_vector_type(16))) _Float16 v16h;
typedef __attribute__((ext_vector_type(8)))  _Float16 v8h;
typedef __attribute__((ext_vector_type(8)))  float    v8f;

#define HCH   28      // hidden channels
#define CP32  32      // padded 28/29 -> 32
#define CP64  64      // padded 56 -> 64
#define NBL   8       // blades in Cl(3,0)
#define NPATH 20

__device__ __constant__ int c_bits[8] = {0,1,2,4,3,5,6,7};  // blade bitmasks
__device__ __constant__ int c_bti[8]  = {0,1,2,4,3,5,6,7};  // bitmask -> blade idx
__device__ __constant__ int c_grade[8]= {0,1,1,1,2,2,2,3};

__device__ inline float signEuclid(int a, int b) {
  a >>= 1; int s = 0;
  while (a) { s += __popc(a & b); a >>= 1; }
  return (s & 1) ? -1.f : 1.f;
}

__device__ inline bool pathsTrue(int gi, int gj, int gk) {
  for (int a = 0; a < 8; ++a) {
    if (__popc(c_bits[a]) != gi) continue;
    for (int c = 0; c < 8; ++c) {
      if (__popc(c_bits[c]) != gk) continue;
      if (__popc(c_bits[a] ^ c_bits[c]) == gj) return true;
    }
  }
  return false;
}

__device__ inline int pathIndex(int gi, int gj, int gk) {
  int idx = 0;
  for (int i = 0; i < 4; ++i)
    for (int j = 0; j < 4; ++j)
      for (int k = 0; k < 4; ++k) {
        bool t = pathsTrue(i, j, k);
        if (i == gi && j == gj && k == gk) return t ? idx : 0;
        if (t) ++idx;
      }
  return 0;
}

__device__ inline v16h mk16(v8h lo, v8h hi) {
  return __builtin_shufflevector(lo, hi, 0,1,2,3,4,5,6,7,8,9,10,11,12,13,14,15);
}

// ---------------------------------------------------------------------------
// Weight conversion: f32 (cout,cin,4) -> f16 [8][32][cinP], B^T layout so a
// WMMA B-fragment load is contiguous per output channel.
// ---------------------------------------------------------------------------
__global__ void clgnn_conv_w(const float* __restrict__ w, _Float16* __restrict__ wt,
                             int cout, int cin, int cinP) {
  int idx = blockIdx.x * blockDim.x + threadIdx.x;
  int total = NBL * CP32 * cinP;
  if (idx >= total) return;
  int i = idx / (CP32 * cinP);
  int r = idx % (CP32 * cinP);
  int n = r / cinP, m = r % cinP;
  float v = 0.f;
  if (n < cout && m < cin) v = w[(n * cin + m) * 4 + c_grade[i]];
  wt[idx] = (_Float16)v;
}

// ---------------------------------------------------------------------------
// Expand gp_w (C,20) into dense per-channel 8x8 signed coefficient table.
// ---------------------------------------------------------------------------
__global__ void clgnn_expand_gp(const float* __restrict__ gp_w,
                                float* __restrict__ coef, int C) {
  int idx = blockIdx.x * blockDim.x + threadIdx.x;
  if (idx >= C * 64) return;
  int c = idx >> 6, i = (idx >> 3) & 7, k = idx & 7;
  int bi = c_bits[i], bk = c_bits[k];
  int j  = c_bti[bi ^ bk];
  int p  = pathIndex(c_grade[i], c_grade[j], c_grade[k]);
  coef[idx] = gp_w[c * NPATH + p] * signEuclid(bi, bk);
}

// ---------------------------------------------------------------------------
// Fully fused CEMLP block.
// grid.x = B/16, block = 256 threads (8 waves); wave w owns blade w.
// ---------------------------------------------------------------------------
__global__ void __launch_bounds__(256)
clgnn_cemlp_block(const _Float16* __restrict__ X,    // [8][B][cinP]
                  const _Float16* __restrict__ Wlin, // [8][32][cinP]
                  const _Float16* __restrict__ Wr,   // [8][32][32]
                  const _Float16* __restrict__ Wl,   // [8][32][32]
                  const float* __restrict__ lin_b,   // [28]
                  const float* __restrict__ left_b,  // [28]
                  const float* __restrict__ sa,      // [28][4] silu_a
                  const float* __restrict__ sb,      // [28][4] silu_b
                  const float* __restrict__ norm_a,  // [28][4]
                  const float* __restrict__ coef,    // [28][64]
                  const float* __restrict__ ln_a,    // [28]
                  _Float16* __restrict__ Out,        // [8][B][32]
                  int B, int cinP) {
  __shared__ __align__(16) _Float16 sA [NBL][16 * CP64]; // async-staged A tiles
  __shared__ __align__(16) float    sY [NBL][16][CP32];  // lin out, reused for right out
  __shared__ __align__(16) _Float16 sX2[NBL][16][CP32];  // silu out (WMMA-A layout)
  __shared__ __align__(16) float    sL [NBL][16][CP32];  // left out

  const int tid  = threadIdx.x;
  const int wave = tid >> 5;     // == blade index
  const int lane = tid & 31;
  const int half = lane >> 4;
  const int l15  = lane & 15;
  const int b0   = blockIdx.x * 16;
  const int gr[8] = {0,1,1,1,2,2,2,3};

  // -------- stage 1a: async copy of this blade's 16 x cinP A tile into LDS ----
  // Tile is 32*cinP bytes; each lane moves 16B per issue (ASYNCcnt-tracked DMA).
  {
    const _Float16* Ab = X + ((size_t)wave * B + b0) * cinP;
    const int chunks = cinP >> 4;  // 16-half chunks per 32 lanes: cinP/16 issues
    for (int it = 0; it < chunks; ++it) {
      const int elem = (it * 32 + lane) * 8;                 // halves, row-major
      unsigned long long gsrc = (unsigned long long)(uintptr_t)(Ab + elem);
      unsigned ldsa = (unsigned)(uintptr_t)(&sA[wave][0]) + (unsigned)(elem * 2);
      asm volatile("global_load_async_to_lds_b128 %0, %1, off"
                   :: "v"(ldsa), "v"(gsrc) : "memory");
    }
  }

  // -------- stage 1b: y = lin(x) : A from LDS (after async wait), B from L2 ---
  {
    const _Float16* Bb = Wlin + (size_t)wave * CP32 * cinP;
    __builtin_prefetch(X + ((size_t)wave * B + b0 + 16) * cinP, 0, 1);
    asm volatile("s_wait_asynccnt 0x0" ::: "memory");
    const _Float16* Asm = &sA[wave][0];
    v8f acc0 = {}, acc1 = {};
    for (int kc = 0; kc < cinP; kc += 32) {
      const int k0 = kc + half * 8;
      v16h a = mk16(*(const v8h*)(Asm + (size_t)l15 * cinP + k0),
                    *(const v8h*)(Asm + (size_t)l15 * cinP + k0 + 16));
      v16h w0 = mk16(*(const v8h*)(Bb + (size_t)l15 * cinP + k0),
                     *(const v8h*)(Bb + (size_t)l15 * cinP + k0 + 16));
      v16h w1 = mk16(*(const v8h*)(Bb + (size_t)(16 + l15) * cinP + k0),
                     *(const v8h*)(Bb + (size_t)(16 + l15) * cinP + k0 + 16));
      acc0 = __builtin_amdgcn_wmma_f32_16x16x32_f16(false, a, false, w0,
                                                    (short)0, acc0, false, false);
      acc1 = __builtin_amdgcn_wmma_f32_16x16x32_f16(false, a, false, w1,
                                                    (short)0, acc1, false, false);
    }
#pragma unroll
    for (int r = 0; r < 8; ++r) {
      sY[wave][r + half * 8][l15]      = acc0[r];
      sY[wave][r + half * 8][16 + l15] = acc1[r];
    }
  }
  __syncthreads();

  // -------- stage 2: MVSiLU (adds lin bias on scalar blade) -------------------
  for (int p = tid; p < 16 * CP32; p += 256) {
    const int b = p >> 5, c = p & 31;
    const bool act = (c < HCH);
    float x[8];
#pragma unroll
    for (int i = 0; i < 8; ++i) x[i] = sY[i][b][c];
    if (act) x[0] += lin_b[c];
    float inv[4];
    inv[0] = x[0];
    inv[1] = x[1]*x[1] + x[2]*x[2] + x[3]*x[3];
    inv[2] = x[4]*x[4] + x[5]*x[5] + x[6]*x[6];
    inv[3] = x[7]*x[7];
    float gate[4];
#pragma unroll
    for (int g = 0; g < 4; ++g) {
      float a  = act ? sa[c * 4 + g] : 0.f;
      float bb = act ? sb[c * 4 + g] : 0.f;
      gate[g] = 1.f / (1.f + expf(-(a * inv[g] + bb)));
    }
#pragma unroll
    for (int i = 0; i < 8; ++i) sX2[i][b][c] = (_Float16)(x[i] * gate[gr[i]]);
  }
  __syncthreads();

  // -------- stage 3: xr = right(x2), xl = left(x2) : A from LDS ---------------
  {
    const _Float16* A2 = &sX2[wave][0][0];
    const int k0 = half * 8;
    v16h a = mk16(*(const v8h*)(A2 + l15 * CP32 + k0),
                  *(const v8h*)(A2 + l15 * CP32 + k0 + 16));
    const _Float16* Br = Wr + (size_t)wave * CP32 * CP32;
    const _Float16* Bl = Wl + (size_t)wave * CP32 * CP32;
    v16h br0 = mk16(*(const v8h*)(Br + (size_t)l15 * CP32 + k0),
                    *(const v8h*)(Br + (size_t)l15 * CP32 + k0 + 16));
    v16h br1 = mk16(*(const v8h*)(Br + (size_t)(16 + l15) * CP32 + k0),
                    *(const v8h*)(Br + (size_t)(16 + l15) * CP32 + k0 + 16));
    v16h bl0 = mk16(*(const v8h*)(Bl + (size_t)l15 * CP32 + k0),
                    *(const v8h*)(Bl + (size_t)l15 * CP32 + k0 + 16));
    v16h bl1 = mk16(*(const v8h*)(Bl + (size_t)(16 + l15) * CP32 + k0),
                    *(const v8h*)(Bl + (size_t)(16 + l15) * CP32 + k0 + 16));
    v8f r0 = {}, r1 = {}, l0 = {}, l1 = {};
    r0 = __builtin_amdgcn_wmma_f32_16x16x32_f16(false, a, false, br0, (short)0, r0, false, false);
    r1 = __builtin_amdgcn_wmma_f32_16x16x32_f16(false, a, false, br1, (short)0, r1, false, false);
    l0 = __builtin_amdgcn_wmma_f32_16x16x32_f16(false, a, false, bl0, (short)0, l0, false, false);
    l1 = __builtin_amdgcn_wmma_f32_16x16x32_f16(false, a, false, bl1, (short)0, l1, false, false);
#pragma unroll
    for (int r = 0; r < 8; ++r) {
      sY[wave][r + half * 8][l15]      = r0[r];   // right result reuses sY
      sY[wave][r + half * 8][16 + l15] = r1[r];
      sL[wave][r + half * 8][l15]      = l0[r];
      sL[wave][r + half * 8][16 + l15] = l1[r];
    }
  }
  __syncthreads();

  // -------- stage 4: normalize(xr) -> weighted GP -> /sqrt2 -> MV layernorm ---
  for (int bb = wave; bb < 16; bb += 8) {
    const int c = lane;
    const bool act = (c < HCH);
    float xr[8], x2[8], y[8];
#pragma unroll
    for (int i = 0; i < 8; ++i) {
      xr[i] = sY[i][bb][c];
      x2[i] = (float)sX2[i][bb][c];
      y[i]  = sL[i][bb][c];
    }
    if (act) y[0] += left_b[c];
    // grade-wise normalization of xr
    float q[4];
    q[0] = xr[0]*xr[0];
    q[1] = xr[1]*xr[1] + xr[2]*xr[2] + xr[3]*xr[3];
    q[2] = xr[4]*xr[4] + xr[5]*xr[5] + xr[6]*xr[6];
    q[3] = xr[7]*xr[7];
    float nrm[4];
#pragma unroll
    for (int g = 0; g < 4; ++g) {
      float s  = powf(q[g]*q[g] + 1e-16f, 0.25f);
      float av = act ? norm_a[c * 4 + g] : 0.f;
      float sg = 1.f / (1.f + expf(-av));
      nrm[g] = sg * (s - 1.f) + 1.f + 1e-6f;
    }
#pragma unroll
    for (int i = 0; i < 8; ++i) xr[i] /= nrm[gr[i]];
    // weighted geometric product (Cayley: one output blade per (i,k) pair)
#pragma unroll
    for (int i = 0; i < 8; ++i) {
#pragma unroll
      for (int k = 0; k < 8; ++k) {
        float cf = act ? coef[((size_t)c << 6) + i * 8 + k] : 0.f;
        int j = c_bti[c_bits[i] ^ c_bits[k]];
        y[j] += cf * x2[i] * xr[k];
      }
    }
    const float inv_sqrt2 = 0.70710678118f;
#pragma unroll
    for (int i = 0; i < 8; ++i) y[i] *= inv_sqrt2;
    // MV layernorm: channel mean of multivector norm (lane = channel)
    float qt = 0.f;
#pragma unroll
    for (int i = 0; i < 8; ++i) qt += y[i] * y[i];
    float s = act ? powf(qt*qt + 1e-16f, 0.25f) : 0.f;
    float sum = s;
#pragma unroll
    for (int off = 16; off > 0; off >>= 1) sum += __shfl_xor(sum, off, 32);
    float mean = sum / (float)HCH + 1e-6f;
    float la = act ? ln_a[c] : 0.f;
#pragma unroll
    for (int i = 0; i < 8; ++i)
      Out[((size_t)i * B + (b0 + bb)) * CP32 + c] = (_Float16)(la * y[i] / mean);
  }
}

// ---------------------------------------------------------------------------
// Graph plumbing kernels
// ---------------------------------------------------------------------------
__global__ void clgnn_zero(float* p, size_t n) {
  size_t i = (size_t)blockIdx.x * blockDim.x + threadIdx.x;
  if (i < n) p[i] = 0.f;
}

__global__ void clgnn_reduce_loc(const float* __restrict__ loc, float* sum, int N) {
  int b = blockIdx.x * blockDim.x + threadIdx.x;
  if (b >= N) return;
  atomicAdd(&sum[0], loc[b*3+0]);
  atomicAdd(&sum[1], loc[b*3+1]);
  atomicAdd(&sum[2], loc[b*3+2]);
}

__global__ void clgnn_embed(const float* __restrict__ loc, const float* __restrict__ vel,
                            const float* __restrict__ chg, const float* __restrict__ locsum,
                            const float* __restrict__ w, const float* __restrict__ bias,
                            _Float16* __restrict__ Hn, int N) {
  int idx = blockIdx.x * blockDim.x + threadIdx.x;
  int b = idx >> 5, n = idx & 31;
  if (b >= N) return;
  float lm[3], vv[3];
#pragma unroll
  for (int d = 0; d < 3; ++d) {
    lm[d] = loc[b*3+d] - locsum[d] / (float)N;
    vv[d] = vel[b*3+d];
  }
  float ch = chg[b];
#pragma unroll
  for (int i = 0; i < 8; ++i) {
    float acc = 0.f;
    if (n < HCH) {
      float x0 = (i == 0) ? ch : 0.f;
      float x1 = (i >= 1 && i <= 3) ? lm[i-1] : 0.f;
      float x2 = (i >= 1 && i <= 3) ? vv[i-1] : 0.f;
      acc = w[n*3+0]*x0 + w[n*3+1]*x1 + w[n*3+2]*x2;
      if (i == 0) acc += bias[n];
    }
    Hn[((size_t)i * N + b) * CP32 + n] = (_Float16)acc;
  }
}

__global__ void clgnn_edge_input(const _Float16* __restrict__ Hn,
                                 const float* __restrict__ chg,
                                 const int* __restrict__ row, const int* __restrict__ col,
                                 _Float16* __restrict__ Xe, int E, int N) {
  int idx = blockIdx.x * blockDim.x + threadIdx.x;
  int e = idx / CP32, c = idx % CP32;
  if (e >= E) return;
  int r = row[e], cl = col[e];
#pragma unroll
  for (int i = 0; i < 8; ++i) {
    float v = 0.f;
    if (c < HCH)
      v = (float)Hn[((size_t)i * N + r) * CP32 + c] -
          (float)Hn[((size_t)i * N + cl) * CP32 + c];
    else if (c == HCH && i == 0)
      v = chg[r] * chg[cl];
    Xe[((size_t)i * E + e) * CP32 + c] = (_Float16)v;
  }
}

__global__ void clgnn_scatter(const _Float16* __restrict__ Msg,
                              const int* __restrict__ row,
                              float* __restrict__ Agg, float* __restrict__ Cnt,
                              int E, int N) {
  int idx = blockIdx.x * blockDim.x + threadIdx.x;
  int e = idx / CP32, c = idx % CP32;
  if (e >= E) return;
  int r = row[e];
#pragma unroll
  for (int i = 0; i < 8; ++i)
    atomicAdd(&Agg[((size_t)i * N + r) * CP32 + c],
              (float)Msg[((size_t)i * E + e) * CP32 + c]);
  if (c == 0) atomicAdd(&Cnt[r], 1.f);
}

__global__ void clgnn_node_input(const _Float16* __restrict__ Hn,
                                 const float* __restrict__ Agg,
                                 const float* __restrict__ Cnt,
                                 _Float16* __restrict__ Xn, int N) {
  int idx = blockIdx.x * blockDim.x + threadIdx.x;
  int b = idx / CP64, c = idx % CP64;
  if (b >= N) return;
  float cnt = fmaxf(Cnt[b], 1.f);
#pragma unroll
  for (int i = 0; i < 8; ++i) {
    float v = 0.f;
    if (c < HCH)           v = (float)Hn[((size_t)i * N + b) * CP32 + c];
    else if (c < 2 * HCH)  v = Agg[((size_t)i * N + b) * CP32 + (c - HCH)] / cnt;
    Xn[((size_t)i * N + b) * CP64 + c] = (_Float16)v;
  }
}

__global__ void clgnn_residual(_Float16* __restrict__ Hn,
                               const _Float16* __restrict__ D, size_t n) {
  size_t i = (size_t)blockIdx.x * blockDim.x + threadIdx.x;
  if (i < n) Hn[i] = (_Float16)((float)Hn[i] + (float)D[i]);
}

__global__ void clgnn_project(const _Float16* __restrict__ Hn,
                              const float* __restrict__ w,
                              const float* __restrict__ loc,
                              float* __restrict__ out, int N) {
  int b = blockIdx.x * blockDim.x + threadIdx.x;
  if (b >= N) return;
#pragma unroll
  for (int d = 0; d < 3; ++d) {
    int i = 1 + d;  // grade-1 blades
    float v = 0.f;
    for (int c = 0; c < HCH; ++c)
      v += w[c * 4 + 1] * (float)Hn[((size_t)i * N + b) * CP32 + c];
    out[b * 3 + d] = loc[b * 3 + d] + v;
  }
}

// ---------------------------------------------------------------------------
// Host orchestration
// ---------------------------------------------------------------------------
struct BlockParams {
  const float *lin_b, *lin_w, *ln_a, *gp_w, *left_b, *left_w, *norm_a, *right_w,
              *silu_a, *silu_b;
};

static BlockParams getBlock(void* const* d_in, int base) {
  BlockParams p;
  p.lin_b   = (const float*)d_in[base + 0];
  p.lin_w   = (const float*)d_in[base + 1];
  p.ln_a    = (const float*)d_in[base + 2];
  p.gp_w    = (const float*)d_in[base + 3];
  p.left_b  = (const float*)d_in[base + 4];
  p.left_w  = (const float*)d_in[base + 5];
  p.norm_a  = (const float*)d_in[base + 6];
  p.right_w = (const float*)d_in[base + 7];
  p.silu_a  = (const float*)d_in[base + 8];
  p.silu_b  = (const float*)d_in[base + 9];
  return p;
}

static void runBlock(hipStream_t s, const BlockParams& p,
                     const _Float16* x, int cin, int cinP, _Float16* out,
                     int B, _Float16* WtLin, _Float16* WtR, _Float16* WtL,
                     float* coef) {
  int t;
  t = NBL * CP32 * cinP;
  clgnn_conv_w<<<(t + 255) / 256, 256, 0, s>>>(p.lin_w,   WtLin, HCH, cin, cinP);
  t = NBL * CP32 * CP32;
  clgnn_conv_w<<<(t + 255) / 256, 256, 0, s>>>(p.right_w, WtR,   HCH, HCH, CP32);
  clgnn_conv_w<<<(t + 255) / 256, 256, 0, s>>>(p.left_w,  WtL,   HCH, HCH, CP32);
  clgnn_expand_gp<<<(HCH * 64 + 255) / 256, 256, 0, s>>>(p.gp_w, coef, HCH);
  clgnn_cemlp_block<<<B / 16, 256, 0, s>>>(x, WtLin, WtR, WtL,
                                           p.lin_b, p.left_b, p.silu_a, p.silu_b,
                                           p.norm_a, coef, p.ln_a, out, B, cinP);
}

extern "C" void kernel_launch(void* const* d_in, const int* in_sizes, int n_in,
                              void* d_out, int out_size, void* d_ws, size_t ws_size,
                              hipStream_t stream) {
  const float* loc = (const float*)d_in[0];
  const float* vel = (const float*)d_in[1];
  const float* chg = (const float*)d_in[2];
  const int*   ei  = (const int*)d_in[3];
  const int N = in_sizes[0] / 3;
  const int E = in_sizes[3] / 2;
  const int* row = ei;
  const int* col = ei + E;

  const float* emb_b  = (const float*)d_in[4];
  const float* emb_w  = (const float*)d_in[5];
  const float* proj_w = (const float*)d_in[87];

  // workspace bump allocator (deterministic per call)
  char* base = (char*)d_ws;
  size_t off = 0;
  auto alloc = [&](size_t bytes) -> void* {
    off = (off + 255) & ~(size_t)255;
    void* p = base + off;
    off += bytes;
    return p;
  };
  _Float16* hN   = (_Float16*)alloc((size_t)NBL * N * CP32 * 2);
  _Float16* eA   = (_Float16*)alloc((size_t)NBL * E * CP32 * 2);
  _Float16* eB   = (_Float16*)alloc((size_t)NBL * E * CP32 * 2);
  _Float16* Xn   = (_Float16*)alloc((size_t)NBL * N * CP64 * 2);
  _Float16* nA   = (_Float16*)alloc((size_t)NBL * N * CP32 * 2);
  _Float16* nB   = (_Float16*)alloc((size_t)NBL * N * CP32 * 2);
  float*    Agg  = (float*)alloc((size_t)NBL * N * CP32 * 4);
  float*    Cnt  = (float*)alloc((size_t)N * 4);
  _Float16* WtLin= (_Float16*)alloc((size_t)NBL * CP32 * CP64 * 2);
  _Float16* WtR  = (_Float16*)alloc((size_t)NBL * CP32 * CP32 * 2);
  _Float16* WtL  = (_Float16*)alloc((size_t)NBL * CP32 * CP32 * 2);
  float*    coef = (float*)alloc((size_t)CP32 * 64 * 4);
  float*    lsum = (float*)alloc(16);

  // ---- embedding ----
  clgnn_zero<<<1, 32, 0, stream>>>(lsum, 4);
  clgnn_reduce_loc<<<(N + 255) / 256, 256, 0, stream>>>(loc, lsum, N);
  clgnn_embed<<<(N * CP32 + 255) / 256, 256, 0, stream>>>(loc, vel, chg, lsum,
                                                          emb_w, emb_b, hN, N);

  // ---- EGCL layers ----
  for (int L = 0; L < 2; ++L) {
    int ebase = 6 + L * 40;   // edge blocks at +0,+10; node blocks at +20,+30
    BlockParams e0 = getBlock(d_in, ebase + 0);
    BlockParams e1 = getBlock(d_in, ebase + 10);
    BlockParams n0 = getBlock(d_in, ebase + 20);
    BlockParams n1 = getBlock(d_in, ebase + 30);

    // edge messages (two fused CEMLP blocks)
    clgnn_edge_input<<<((size_t)E * CP32 + 255) / 256, 256, 0, stream>>>(
        hN, chg, row, col, eA, E, N);
    runBlock(stream, e0, eA, HCH + 1, CP32, eB, E, WtLin, WtR, WtL, coef);
    runBlock(stream, e1, eB, HCH,     CP32, eA, E, WtLin, WtR, WtL, coef);

    // mean aggregation by destination node
    clgnn_zero<<<((size_t)NBL * N * CP32 + 255) / 256, 256, 0, stream>>>(
        Agg, (size_t)NBL * N * CP32);
    clgnn_zero<<<(N + 255) / 256, 256, 0, stream>>>(Cnt, (size_t)N);
    clgnn_scatter<<<((size_t)E * CP32 + 255) / 256, 256, 0, stream>>>(
        eA, row, Agg, Cnt, E, N);

    // node update (two fused CEMLP blocks) + residual
    clgnn_node_input<<<((size_t)N * CP64 + 255) / 256, 256, 0, stream>>>(
        hN, Agg, Cnt, Xn, N);
    runBlock(stream, n0, Xn, 2 * HCH, CP64, nA, N, WtLin, WtR, WtL, coef);
    runBlock(stream, n1, nA, HCH,     CP32, nB, N, WtLin, WtR, WtL, coef);
    clgnn_residual<<<((size_t)NBL * N * CP32 + 255) / 256, 256, 0, stream>>>(
        hN, nB, (size_t)NBL * N * CP32);
  }

  // ---- projection + final position update ----
  clgnn_project<<<(N + 255) / 256, 256, 0, stream>>>(hN, proj_w, loc, (float*)d_out, N);
  (void)n_in; (void)out_size; (void)ws_size;
}
